// TopkSparseAutoEncoder_7456063225987
// MI455X (gfx1250) — compile-verified, matrix operand order checked by reading, not run on hardware
//
#include <hip/hip_runtime.h>
#include <hip/hip_bf16.h>

#define SEQ     8192
#define LLMD    768
#define SAEH    24576
#define TOPK    150
#define NBINS   512
#define CAND_CAP 2048

typedef __attribute__((ext_vector_type(16))) __bf16 v16bf;
typedef __attribute__((ext_vector_type(8)))  __bf16 v8bf;
typedef __attribute__((ext_vector_type(8)))  float  v8f;

__device__ __forceinline__ unsigned mono_key(float v) {
    unsigned u = __float_as_uint(v);
    return (u & 0x80000000u) ? ~u : (u | 0x80000000u);
}

__device__ __forceinline__ v16bf load_frag(const __bf16* row, int k0) {
    v8bf lo = *(const v8bf*)(row + k0);
    v8bf hi = *(const v8bf*)(row + k0 + 16);
    return __builtin_shufflevector(lo, hi, 0,1,2,3,4,5,6,7,8,9,10,11,12,13,14,15);
}

// ---------------- fp32 -> bf16 (RNE) ----------------
__global__ __launch_bounds__(256) void tobf16_kernel(const float* __restrict__ in,
                                                     unsigned short* __restrict__ out,
                                                     long long n) {
    long long i = (long long)blockIdx.x * 256 + threadIdx.x;
    long long stride = (long long)gridDim.x * 256;
    for (; i < n; i += stride) {
        unsigned u = __float_as_uint(in[i]);
        unsigned short b;
        if ((u & 0x7fffffffu) > 0x7f800000u) b = (unsigned short)((u >> 16) | 0x40u); // NaN
        else b = (unsigned short)((u + 0x7fffu + ((u >> 16) & 1u)) >> 16);
        out[i] = b;
    }
}

// ---------------- zero u32 buffer ----------------
__global__ __launch_bounds__(256) void zero_kernel(unsigned* __restrict__ p, long long n) {
    long long i = (long long)blockIdx.x * 256 + threadIdx.x;
    long long stride = (long long)gridDim.x * 256;
    for (; i < n; i += stride) p[i] = 0u;
}

// ---------------- W_dec [768, 24576] -> WdecT [24576, 768] ----------------
__global__ __launch_bounds__(256) void transpose_kernel(const float* __restrict__ in,
                                                        float* __restrict__ out) {
    __shared__ float tile[32][33];
    int tx = threadIdx.x & 31;
    int ty = threadIdx.x >> 5;          // 0..7
    int hBase = blockIdx.x * 32;        // SAE dim
    int dBase = blockIdx.y * 32;        // LLM dim
#pragma unroll
    for (int i = 0; i < 4; ++i) {
        int d = dBase + ty + i * 8;
        tile[ty + i * 8][tx] = in[(long long)d * SAEH + hBase + tx];
    }
    __syncthreads();
#pragma unroll
    for (int i = 0; i < 4; ++i) {
        int h = hBase + ty + i * 8;
        out[(long long)h * LLMD + dBase + tx] = tile[tx][ty + i * 8];
    }
}

// ---------------- encode GEMM (bf16 WMMA, 32x64 per wave) + per-row radix histogram ----
// grid: (SAEH/512, SEQ/32), block 256 (8 waves).
// wave w owns M-rows [by*32, by*32+32) x N-cols [bx*512 + w*64, ... +64):
//   2 M-tiles x 4 N-tiles = 8 WMMA per K-step; A frags reused 4x, B frags reused 2x.
__global__ __launch_bounds__(256) void encode_kernel(const __bf16* __restrict__ Xb,
                                                     const __bf16* __restrict__ Wb,
                                                     const float* __restrict__ b_enc,
                                                     float* __restrict__ pre,
                                                     unsigned* __restrict__ ghist) {
    __shared__ unsigned lhist[32 * NBINS];          // 64 KB
    for (int i = threadIdx.x; i < 32 * NBINS; i += 256) lhist[i] = 0u;
    __syncthreads();

    const int wave = threadIdx.x >> 5;
    const int lane = threadIdx.x & 31;
    const int half = lane >> 4;         // 0 or 1
    const int m16  = lane & 15;
    const int mBase = blockIdx.y * 32;
    const int nBase = blockIdx.x * 512 + wave * 64;

    const __bf16* arow0 = Xb + (long long)(mBase + m16) * LLMD;
    const __bf16* arow1 = arow0 + 16LL * LLMD;
    const __bf16* brow[4];
#pragma unroll
    for (int nt = 0; nt < 4; ++nt)
        brow[nt] = Wb + (long long)(nBase + nt * 16 + m16) * LLMD;

    v8f c[2][4];
#pragma unroll
    for (int mt = 0; mt < 2; ++mt)
#pragma unroll
        for (int nt = 0; nt < 4; ++nt)
            c[mt][nt] = (v8f){};

#pragma unroll 2
    for (int k = 0; k < LLMD; k += 32) {
        const int k0 = k + half * 8;     // bf16 16x32 frag: lanes<16 K={0..7,16..23}, lanes>=16 K={8..15,24..31}
        v16bf A0 = load_frag(arow0, k0);
        v16bf A1 = load_frag(arow1, k0);
        v16bf B0 = load_frag(brow[0], k0);
        v16bf B1 = load_frag(brow[1], k0);
        v16bf B2 = load_frag(brow[2], k0);
        v16bf B3 = load_frag(brow[3], k0);
        if (k + 32 < LLMD) {
            __builtin_prefetch(arow0 + k0 + 32, 0, 3);
            __builtin_prefetch(brow[0] + k0 + 32, 0, 3);
            __builtin_prefetch(brow[2] + k0 + 32, 0, 3);
        }
        c[0][0] = __builtin_amdgcn_wmma_f32_16x16x32_bf16(false, A0, false, B0, (short)0, c[0][0], false, false);
        c[0][1] = __builtin_amdgcn_wmma_f32_16x16x32_bf16(false, A0, false, B1, (short)0, c[0][1], false, false);
        c[0][2] = __builtin_amdgcn_wmma_f32_16x16x32_bf16(false, A0, false, B2, (short)0, c[0][2], false, false);
        c[0][3] = __builtin_amdgcn_wmma_f32_16x16x32_bf16(false, A0, false, B3, (short)0, c[0][3], false, false);
        c[1][0] = __builtin_amdgcn_wmma_f32_16x16x32_bf16(false, A1, false, B0, (short)0, c[1][0], false, false);
        c[1][1] = __builtin_amdgcn_wmma_f32_16x16x32_bf16(false, A1, false, B1, (short)0, c[1][1], false, false);
        c[1][2] = __builtin_amdgcn_wmma_f32_16x16x32_bf16(false, A1, false, B2, (short)0, c[1][2], false, false);
        c[1][3] = __builtin_amdgcn_wmma_f32_16x16x32_bf16(false, A1, false, B3, (short)0, c[1][3], false, false);
    }

#pragma unroll
    for (int nt = 0; nt < 4; ++nt) {
        const int col  = nBase + nt * 16 + m16;
        const float bias = b_enc[col];
#pragma unroll
        for (int mt = 0; mt < 2; ++mt) {
#pragma unroll
            for (int i = 0; i < 8; ++i) {
                float v = c[mt][nt][i] + bias;
                int r = mt * 16 + i + half * 8;              // local row 0..31
                long long gi = (long long)(mBase + r) * SAEH + col;
                __builtin_nontemporal_store(v, pre + gi);    // 805 MB stream
                unsigned bin = mono_key(v) >> 23;            // sign+exponent radix
                atomicAdd(&lhist[r * NBINS + bin], 1u);
            }
        }
    }
    __syncthreads();

    for (int i = threadIdx.x; i < 32 * NBINS; i += 256) {
        unsigned cnt = lhist[i];
        if (cnt) atomicAdd(&ghist[(long long)(mBase + (i >> 9)) * NBINS + (i & (NBINS - 1))], cnt);
    }
}

// ---------------- per-row coarse threshold from histogram ----------------
__global__ __launch_bounds__(256) void threshold_kernel(const unsigned* __restrict__ ghist,
                                                        int* __restrict__ thrbin,
                                                        int* __restrict__ takecnt) {
    int row = blockIdx.x * 256 + threadIdx.x;
    if (row >= SEQ) return;
    const unsigned* h = ghist + (long long)row * NBINS;
    int acc = 0, bin = NBINS - 1;
    for (; bin >= 0; --bin) {
        int c = (int)h[bin];
        if (acc + c >= TOPK) break;
        acc += c;
    }
    thrbin[row]  = bin;          // values in bins > bin are all kept (acc < TOPK of them)
    takecnt[row] = TOPK - acc;   // how many to take from the boundary bin
}

// ---------------- scan row, exact tie-select in boundary bin, sparse decode ----------------
__global__ __launch_bounds__(256) void topk_decode_kernel(const float* __restrict__ pre,
                                                          const int* __restrict__ thrbin,
                                                          const int* __restrict__ takecnt,
                                                          const float* __restrict__ WdecT,
                                                          const float* __restrict__ b_dec,
                                                          float* __restrict__ out) {
    __shared__ int   s_idx[160];
    __shared__ float s_val[160];
    __shared__ int   c_idx[CAND_CAP];
    __shared__ float c_val[CAND_CAP];
    __shared__ float r_val[256];
    __shared__ int   r_pos[256];
    __shared__ int   n_sel, n_cand;

    const int row  = blockIdx.x;
    const int tb   = thrbin[row];
    const int take = takecnt[row];
    if (threadIdx.x == 0) { n_sel = 0; n_cand = 0; }
    __syncthreads();

    const float* prow = pre + (long long)row * SAEH;
    for (int j = threadIdx.x; j < SAEH; j += 256) {
        float v = __builtin_nontemporal_load(prow + j);
        int bin = (int)(mono_key(v) >> 23);
        if (bin > tb) {
            int p = atomicAdd(&n_sel, 1);
            if (p < 160) { s_idx[p] = j; s_val[p] = v; }
        } else if (bin == tb) {
            int p = atomicAdd(&n_cand, 1);
            if (p < CAND_CAP) { c_idx[p] = j; c_val[p] = v; }
        }
    }
    __syncthreads();

    const int base = min(n_sel, 160);
    const int nc   = min(n_cand, CAND_CAP);
    const int need = min(take, nc);

    // iterative arg-max to pull exactly `need` largest from the boundary bin
    for (int t = 0; t < need; ++t) {
        float bv = -3.0e38f; int bp = -1;
        for (int j = threadIdx.x; j < nc; j += 256) {
            float v = c_val[j];
            if (v > bv) { bv = v; bp = j; }
        }
        r_val[threadIdx.x] = bv; r_pos[threadIdx.x] = bp;
        __syncthreads();
        for (int s = 128; s > 0; s >>= 1) {
            if (threadIdx.x < s && r_val[threadIdx.x + s] > r_val[threadIdx.x]) {
                r_val[threadIdx.x] = r_val[threadIdx.x + s];
                r_pos[threadIdx.x] = r_pos[threadIdx.x + s];
            }
            __syncthreads();
        }
        if (threadIdx.x == 0) {
            int p = r_pos[0];
            if (p >= 0 && base + t < 160) {
                s_idx[base + t] = c_idx[p];
                s_val[base + t] = c_val[p];
                c_val[p] = -3.0e38f;
            }
        }
        __syncthreads();
    }
    const int total = min(base + need, 160);   // == TOPK in the common case

    // sparse decode: out[row,d] = b_dec[d] + sum_k val_k * WdecT[idx_k, d]
    for (int d = threadIdx.x; d < LLMD; d += 256) {
        float acc = b_dec[d];
        for (int k = 0; k < total; ++k)
            acc += s_val[k] * WdecT[(long long)s_idx[k] * LLMD + d];   // L2-resident
        out[(long long)row * LLMD + d] = acc;
    }
}

extern "C" void kernel_launch(void* const* d_in, const int* in_sizes, int n_in,
                              void* d_out, int out_size, void* d_ws, size_t ws_size,
                              hipStream_t stream) {
    const float* x     = (const float*)d_in[0];   // [1,8192,768]
    const float* W_enc = (const float*)d_in[1];   // [24576,768]
    const float* b_enc = (const float*)d_in[2];   // [24576]
    const float* W_dec = (const float*)d_in[3];   // [768,24576]
    const float* b_dec = (const float*)d_in[4];   // [768]
    float* out = (float*)d_out;

    char* ws = (char*)d_ws;
    size_t off = 0;
    auto carve = [&](size_t bytes) -> void* {
        void* p = ws + off;
        off = (off + bytes + 255) & ~(size_t)255;
        return p;
    };
    unsigned short* Xb    = (unsigned short*)carve((size_t)SEQ * LLMD * 2);
    unsigned short* Wb    = (unsigned short*)carve((size_t)SAEH * LLMD * 2);
    float*          WdecT = (float*)carve((size_t)SAEH * LLMD * 4);
    float*          pre   = (float*)carve((size_t)SEQ * SAEH * 4);
    unsigned*       ghist = (unsigned*)carve((size_t)SEQ * NBINS * 4);
    int*            thrb  = (int*)carve((size_t)SEQ * 4);
    int*            takec = (int*)carve((size_t)SEQ * 4);

    // 1) convert to bf16
    tobf16_kernel<<<1024, 256, 0, stream>>>(x, Xb, (long long)SEQ * LLMD);
    tobf16_kernel<<<2048, 256, 0, stream>>>(W_enc, Wb, (long long)SAEH * LLMD);
    // 2) transpose decoder weights so feature vectors are contiguous
    transpose_kernel<<<dim3(SAEH / 32, LLMD / 32), 256, 0, stream>>>(W_dec, WdecT);
    // 3) zero global histograms
    zero_kernel<<<2048, 256, 0, stream>>>(ghist, (long long)SEQ * NBINS);
    // 4) encode GEMM (WMMA bf16, 32x64 per wave) + fused per-row histogram
    encode_kernel<<<dim3(SAEH / 512, SEQ / 32), 256, 0, stream>>>(
        (const __bf16*)Xb, (const __bf16*)Wb, b_enc, pre, ghist);
    // 5) per-row coarse top-k threshold
    threshold_kernel<<<SEQ / 256, 256, 0, stream>>>(ghist, thrb, takec);
    // 6) exact selection + sparse decode
    topk_decode_kernel<<<SEQ, 256, 0, stream>>>(pre, thrb, takec, WdecT, b_dec, out);
}
